// MyModel_87522843558812
// MI455X (gfx1250) — matrix operand order repaired and checked by reading, not compiled, not run on hardware
//
#include <hip/hip_runtime.h>
#include <stdint.h>

// Problem constants (from reference): B=32, T=512, V=32000, D=300, U=300, NT=3
#define Bc 32
#define Tc 512
#define Dc 300
#define Uc 300
#define NTc 3
#define Mrows (Bc * Tc)        // 16384 rows of x
#define KPAD 320               // 300 padded to multiple of 32
#define KKc (KPAD / 32)        // 10 k-steps of 32
#define NZ 2400                // 4U forward + 4U backward columns of zx
#define TN_DIR 75              // 1200/16 N-tiles per direction
#define TN_ALL 150             // both directions
#define TM_ALL (Mrows / 16)    // 1024 M-tiles
#define NGRP 19                // ceil(150 / 8) tileN groups per block row

#define HAVE_ASYNC_LDS __has_builtin(__builtin_amdgcn_global_load_async_to_lds_b128)

typedef __attribute__((ext_vector_type(16))) __bf16 v16bf;
typedef __attribute__((ext_vector_type(8)))  float  v8f;
typedef int v4i32 __attribute__((vector_size(16)));   // matches builtin param type

union Frag {
  uint32_t u[8];
  uint4    q[2];
  v16bf    bf;
};

__device__ __forceinline__ uint32_t f32_to_bf16(float f) {
  union { float f; uint32_t u; } c; c.f = f;
  uint32_t u = c.u;
  uint32_t r = u + 0x7FFFu + ((u >> 16) & 1u);   // round-to-nearest-even
  return r >> 16;
}
__device__ __forceinline__ float bf16_to_f32(uint16_t h) {
  union { uint32_t u; float f; } c; c.u = ((uint32_t)h) << 16;
  return c.f;
}
__device__ __forceinline__ float sigm(float x) { return 1.0f / (1.0f + __expf(-x)); }

// ---------------------------------------------------------------------------
// Kernel 1: embedding gather + pack x into WMMA A-fragment layout (bf16)
// xpack[((tileM*KK + kk)*32 + lane)*8 + j] : 16-bit A 16x32 layout per ISA
// ---------------------------------------------------------------------------
__global__ void embed_pack_kernel(const int* __restrict__ inputs,
                                  const float* __restrict__ emb,
                                  uint32_t* __restrict__ xpack) {
  int s = blockIdx.x * blockDim.x + threadIdx.x;   // one packed dword per thread
  int j    = s & 7;
  int s2   = s >> 3;
  int lane = s2 & 31;
  int s3   = s2 >> 5;
  int kk    = s3 % KKc;
  int tileM = s3 / KKc;
  int m  = tileM * 16 + (lane & 15);               // row = b*T + t
  int hi = lane >> 4;
  int kl = ((j >> 2) << 4) + hi * 8 + ((j & 3) << 1);
  int k  = kk * 32 + kl;
  int tok = inputs[m];
  const float* row = emb + (size_t)tok * Dc;
  uint32_t lo = (k     < Dc) ? f32_to_bf16(row[k])     : 0u;
  uint32_t up = (k + 1 < Dc) ? f32_to_bf16(row[k + 1]) : 0u;
  xpack[s] = lo | (up << 16);
}

// ---------------------------------------------------------------------------
// Kernel 2: pack Wk_f|Wk_b and Wr_f|Wr_b into WMMA B-fragment layout (bf16)
// B 32x16 layout: lane = N%16 + 16*h; VGPR j holds K = 16*(j/4) + 8h + 2*(j%4) and +1
// ---------------------------------------------------------------------------
__global__ void pack_weights_kernel(const float* __restrict__ WkF,
                                    const float* __restrict__ WkB,
                                    const float* __restrict__ WrF,
                                    const float* __restrict__ WrB,
                                    uint32_t* __restrict__ wkpack,
                                    uint32_t* __restrict__ wrpack) {
  int s = blockIdx.x * blockDim.x + threadIdx.x;   // < TN_ALL*KKc*32*8 = 384000
  int j    = s & 7;
  int s2   = s >> 3;
  int lane = s2 & 31;
  int s3   = s2 >> 5;
  int kk    = s3 % KKc;
  int tileN = s3 / KKc;
  int nl = lane & 15;
  int hi = lane >> 4;
  int k  = kk * 32 + ((j >> 2) << 4) + hi * 8 + ((j & 3) << 1);
  int n;
  const float *wk, *wr;
  if (tileN < TN_DIR) { n = tileN * 16 + nl;            wk = WkF; wr = WrF; }
  else                { n = (tileN - TN_DIR) * 16 + nl; wk = WkB; wr = WrB; }
  uint32_t a0 = (k     < Dc) ? f32_to_bf16(wk[(size_t)k * 1200 + n])       : 0u;
  uint32_t a1 = (k + 1 < Dc) ? f32_to_bf16(wk[(size_t)(k + 1) * 1200 + n]) : 0u;
  wkpack[s] = a0 | (a1 << 16);
  uint32_t b0 = (k     < Uc) ? f32_to_bf16(wr[(size_t)k * 1200 + n])       : 0u;
  uint32_t b1 = (k + 1 < Uc) ? f32_to_bf16(wr[(size_t)(k + 1) * 1200 + n]) : 0u;
  wrpack[s] = b0 | (b1 << 16);
}

// ---------------------------------------------------------------------------
// Kernel 3: zx = x @ [Wk_f | Wk_b] + [b_f | b_b]   (16384 x 320 x 2400, bf16 WMMA)
// 8 waves per block share one tileM: A fragments staged once into LDS via
// GLOBAL_LOAD_ASYNC_TO_LDS_B128 (8x less redundant L2 read traffic), then each
// wave runs one 16x16 N-tile with 10 k-steps of v_wmma_f32_16x16x32_bf16.
// ---------------------------------------------------------------------------
__global__ __launch_bounds__(256) void gemm_zx_kernel(
    const uint32_t* __restrict__ xpack,
    const uint32_t* __restrict__ wkpack,
    const float* __restrict__ bF,
    const float* __restrict__ bB,
    float* __restrict__ zx) {
  __shared__ __align__(16) uint32_t aS[KKc * 32 * 8];   // 10 KB: this tileM's A frags
  const int tid   = threadIdx.x;
  const int wave  = tid >> 5;
  const int lane  = tid & 31;
  const int tileM = blockIdx.x / NGRP;
  const int grp   = blockIdx.x % NGRP;
  const int tileN = grp * 8 + wave;

  // ---- stage A fragments for this tileM (10 kk x 32 lanes x 32B) into LDS ----
  for (int c = tid; c < KKc * 64; c += 256) {           // 640 16-byte chunks
    const int kk   = c >> 6;
    const int rem  = c & 63;
    const int lsrc = rem >> 1;
    const int half = rem & 1;
    const size_t gdw = ((size_t)(tileM * KKc + kk) * 32 + lsrc) * 8 + half * 4;
    const int    ldw = (kk * 32 + lsrc) * 8 + half * 4;
#if HAVE_ASYNC_LDS
    __builtin_amdgcn_global_load_async_to_lds_b128(
        (__attribute__((address_space(1))) v4i32*)(xpack + gdw),
        (__attribute__((address_space(3))) v4i32*)(&aS[ldw]), 0, 0);
#else
    uint4 v = *(const uint4*)(xpack + gdw);
    *(uint4*)(&aS[ldw]) = v;
#endif
  }
#if HAVE_ASYNC_LDS
#if __has_builtin(__builtin_amdgcn_s_wait_asynccnt)
  __builtin_amdgcn_s_wait_asynccnt(0);
#else
  asm volatile("s_wait_asynccnt 0x0" ::: "memory");
#endif
#endif
  __syncthreads();

  if (tileN < TN_ALL) {
    const int nl = lane & 15;
    const int hi = lane >> 4;
    const int col = tileN * 16 + nl;
    const float bias = (col < 1200) ? bF[col] : bB[col - 1200];
    v8f acc;
#pragma unroll
    for (int r = 0; r < 8; ++r) acc[r] = bias;

    const uint4* aS4 = (const uint4*)aS;
    const uint4* pb  = (const uint4*)wkpack + ((size_t)(tileN * KKc) * 32 + lane) * 2;
#pragma unroll
    for (int kk = 0; kk < KKc; ++kk) {
      Frag fa, fb;
      fa.q[0] = aS4[(kk * 32 + lane) * 2];
      fa.q[1] = aS4[(kk * 32 + lane) * 2 + 1];
      fb.q[0] = pb[0]; fb.q[1] = pb[1];
      pb += 64;                               // next k-step: 32 lanes * 2 uint4
      __builtin_prefetch(pb, 0, 1);           // global_prefetch_b8 on weight stream
      acc = __builtin_amdgcn_wmma_f32_16x16x32_bf16(false, fa.bf, false, fb.bf,
                                                    (short)0, acc, false, false);
    }
    float* out = zx + (size_t)(tileM * 16 + hi * 8) * NZ + col;
#pragma unroll
    for (int r = 0; r < 8; ++r) out[(size_t)r * NZ] = acc[r];
  }
}

// ---------------------------------------------------------------------------
// Kernel 4: LSTM scan. One 32-wave workgroup per direction (grid.x = 2).
// Per step: z = zx[:,t] + h @ Wr via WMMA (2x75 tiles, 10 k-steps), then gates.
// h lives in LDS as bf16 rows padded to KPAD; A-fragments are exactly two
// aligned ds_load_b128 per (row-tile, k-step).
// ---------------------------------------------------------------------------
__global__ __launch_bounds__(1024) void lstm_scan_kernel(
    const float* __restrict__ zx,
    const uint32_t* __restrict__ wrpack,
    const int* __restrict__ seqlen,
    float* __restrict__ hcat) {
  __shared__ __align__(16) uint16_t zS[Bc * 1200];   // 76.8 KB gate pre-activations
  __shared__ __align__(16) uint16_t hS[Bc * KPAD];   // 20.5 KB hidden state (padded)

  const int dir  = blockIdx.x;
  const int tid  = threadIdx.x;
  const int wave = tid >> 5;
  const int lane = tid & 31;
  const int nl = lane & 15;
  const int hi = lane >> 4;

  float c_reg[10], h_reg[10];
#pragma unroll
  for (int i = 0; i < 10; ++i) { c_reg[i] = 0.0f; h_reg[i] = 0.0f; }
  for (int i = tid; i < Bc * KPAD; i += 1024) hS[i] = 0;
  __syncthreads();

  const uint4* hS4 = (const uint4*)hS;               // 8 bf16 per uint4

  for (int s = 0; s < Tc; ++s) {
    const int t = dir ? (Tc - 1 - s) : s;

    // ---- phase 1: z = zx + h @ Wr (bf16 WMMA, acc seeded with zx) ----
    for (int tile = wave; tile < 2 * TN_DIR; tile += 32) {
      const int tileM = tile / TN_DIR;               // batch half: rows 0-15 / 16-31
      const int tn    = tile % TN_DIR;
      const int col   = tn * 16 + nl;                // 0..1199
      v8f acc;
#pragma unroll
      for (int r = 0; r < 8; ++r) {
        const int b = tileM * 16 + hi * 8 + r;
        acc[r] = zx[(size_t)(b * Tc + t) * NZ + dir * 1200 + col];
      }
      const int mrow = tileM * 16 + nl;
      const int abase = mrow * (KPAD / 8) + hi;      // uint4 index: mrow*40 + kk*4 + hi
      const uint4* pb = (const uint4*)wrpack +
                        ((size_t)((dir * TN_DIR + tn) * KKc) * 32 + lane) * 2;
#pragma unroll
      for (int kk = 0; kk < KKc; ++kk) {
        Frag fa, fb;
        fa.q[0] = hS4[abase + kk * 4];               // K = kk*32 + hi*8 + 0..7
        fa.q[1] = hS4[abase + kk * 4 + 2];           // K = kk*32 + 16 + hi*8 + 0..7
        fb.q[0] = pb[0]; fb.q[1] = pb[1];
        pb += 64;
        acc = __builtin_amdgcn_wmma_f32_16x16x32_bf16(false, fa.bf, false, fb.bf,
                                                      (short)0, acc, false, false);
      }
#pragma unroll
      for (int r = 0; r < 8; ++r) {
        const int b = tileM * 16 + hi * 8 + r;
        zS[b * 1200 + col] = (uint16_t)f32_to_bf16(acc[r]);
      }
    }
    __syncthreads();

    // ---- warm L2/L0 for next step's zx rows while gate math runs ----
    if (s + 1 < Tc) {
      const int tn2 = dir ? (t - 1) : (t + 1);
      // 32 rows x 1200 floats = 32 x 38 lines of 128B
      if (tid < Bc * 38) {
        const int b    = tid / 38;
        const int line = tid % 38;
        __builtin_prefetch(&zx[(size_t)(b * Tc + tn2) * NZ + dir * 1200 + line * 32], 0, 1);
      }
    }

    // ---- phase 2: gates + state update (f32 VALU, state in registers) ----
    int it = 0;
    for (int idx = tid; idx < Bc * Uc; idx += 1024, ++it) {
      const int b = idx / Uc;
      const int u = idx - b * Uc;
      const float zi = bf16_to_f32(zS[b * 1200 + u]);
      const float zf = bf16_to_f32(zS[b * 1200 + 300 + u]);
      const float zg = bf16_to_f32(zS[b * 1200 + 600 + u]);
      const float zo = bf16_to_f32(zS[b * 1200 + 900 + u]);
      const float co = c_reg[it], ho = h_reg[it];
      const float cn = sigm(zf) * co + sigm(zi) * tanhf(zg);
      const float hn = sigm(zo) * tanhf(cn);
      const bool  mt = t < seqlen[b];
      const float c2 = mt ? cn : co;
      const float h2 = mt ? hn : ho;
      c_reg[it] = c2;
      h_reg[it] = h2;
      hS[b * KPAD + u] = (uint16_t)f32_to_bf16(h2);
      hcat[(size_t)(b * Tc + t) * 600 + dir * 300 + u] = h2;
    }
    __syncthreads();
  }
}

// ---------------------------------------------------------------------------
// Kernel 5: out = hcat @ fc_W + fc_b   (16384 x 600 x 3)
// One thread per row; float4-vectorized on both operands.
// ---------------------------------------------------------------------------
__global__ void fc_kernel(const float* __restrict__ hcat,
                          const float* __restrict__ fcW,
                          const float* __restrict__ fcb,
                          float* __restrict__ out) {
  int m = blockIdx.x * blockDim.x + threadIdx.x;     // < Mrows
  const float4* hrow = (const float4*)(hcat + (size_t)m * 600);
  float a0 = 0.f, a1 = 0.f, a2 = 0.f;
  for (int q = 0; q < 150; ++q) {                    // 4 rows of fcW per iteration
    const float4 h4 = hrow[q];
    const float4 w0 = *(const float4*)(fcW + q * 12);      // u=4q   : w[0..2], u=4q+1: w[3]
    const float4 w1 = *(const float4*)(fcW + q * 12 + 4);
    const float4 w2 = *(const float4*)(fcW + q * 12 + 8);
    a0 += h4.x * w0.x + h4.y * w0.w + h4.z * w1.z + h4.w * w2.y;
    a1 += h4.x * w0.y + h4.y * w1.x + h4.z * w1.w + h4.w * w2.z;
    a2 += h4.x * w0.z + h4.y * w1.y + h4.z * w2.x + h4.w * w2.w;
  }
  out[(size_t)m * 3 + 0] = a0 + fcb[0];
  out[(size_t)m * 3 + 1] = a1 + fcb[1];
  out[(size_t)m * 3 + 2] = a2 + fcb[2];
}

// ---------------------------------------------------------------------------
extern "C" void kernel_launch(void* const* d_in, const int* in_sizes, int n_in,
                              void* d_out, int out_size, void* d_ws, size_t ws_size,
                              hipStream_t stream) {
  const int*   inputs = (const int*)d_in[0];
  const int*   seqlen = (const int*)d_in[1];
  const float* emb    = (const float*)d_in[2];
  const float* WkF    = (const float*)d_in[3];
  const float* WrF    = (const float*)d_in[4];
  const float* bF     = (const float*)d_in[5];
  const float* WkB    = (const float*)d_in[6];
  const float* WrB    = (const float*)d_in[7];
  const float* bB     = (const float*)d_in[8];
  const float* fcW    = (const float*)d_in[9];
  const float* fcb    = (const float*)d_in[10];
  float* out = (float*)d_out;

  // Workspace layout (all 256B aligned):
  char* ws = (char*)d_ws;
  uint32_t* xpack  = (uint32_t*)(ws);                               // 10,485,760 B
  uint32_t* wkpack = (uint32_t*)(ws + 10485760);                    //  1,536,000 B
  uint32_t* wrpack = (uint32_t*)(ws + 10485760 + 1536000);          //  1,536,000 B
  float*    zx     = (float*)(ws + 10485760 + 2 * 1536000);         // 157,286,400 B
  float*    hcat   = (float*)(ws + 10485760 + 2 * 1536000 + 157286400); // 39,321,600 B

  embed_pack_kernel  <<<10240, 256, 0, stream>>>(inputs, emb, xpack);
  pack_weights_kernel<<< 1500, 256, 0, stream>>>(WkF, WkB, WrF, WrB, wkpack, wrpack);
  gemm_zx_kernel     <<<TM_ALL * NGRP, 256, 0, stream>>>(xpack, wkpack, bF, bB, zx);
  lstm_scan_kernel   <<<    2, 1024, 0, stream>>>(zx, wrpack, seqlen, hcat);
  fc_kernel          <<<Mrows / 256, 256, 0, stream>>>(hcat, fcW, fcb, out);
}
